// MultiHeadAttention_49718541418667
// MI455X (gfx1250) — matrix operand order repaired and checked by reading, not compiled, not run on hardware
//
#include <hip/hip_runtime.h>
#include <hip/hip_bf16.h>

// MHA forward for MI455X (gfx1250, wave32, WMMA + TDM).
// 5 launches: 3x projection GEMM (fp32 in -> f16 head-split out, WMMA f16),
// flash-attention (TDM-staged K/Q tiles, WMMA f16 scores + PV, fp32 online softmax),
// 1x output GEMM (f16 in -> fp32 out + bias).
// Workspace use: 64 MiB of d_ws (q16/k16/v16/ctx16), fully rewritten per call.

typedef _Float16 h8  __attribute__((ext_vector_type(8)));
typedef _Float16 h16 __attribute__((ext_vector_type(16)));
typedef float    f8  __attribute__((ext_vector_type(8)));

typedef unsigned int u32x4 __attribute__((ext_vector_type(4)));
typedef int          i32x4 __attribute__((ext_vector_type(4)));
typedef int          i32x8 __attribute__((ext_vector_type(8)));

#define DMODEL 1024
#define SEQ    2048
#define BATCH  4
#define HEADS  16
#define DK     64
#define MTOT   (BATCH * SEQ)
#define NEG_INF (-1.0e9f)

#if defined(__HIP_DEVICE_COMPILE__) && __has_builtin(__builtin_amdgcn_tensor_load_to_lds) && \
    __has_builtin(__builtin_amdgcn_s_wait_tensorcnt)
#define HAVE_TDM 1
#else
#define HAVE_TDM 0
#endif

__device__ __forceinline__ f8 wmma_16x16x32(h16 a, h16 b, f8 c) {
  // D = A(16x32 f16) * B(32x16 f16) + C(16x16 f32)
  return __builtin_amdgcn_wmma_f32_16x16x32_f16(false, a, false, b, (short)0, c,
                                                false, false);
}

#if HAVE_TDM
// 1-D contiguous f16 copy global -> LDS via the Tensor Data Mover.
// D# per CDNA5 ISA: group0 = {count|flags, lds_addr,
// global_addr[31:0], global_addr[56:32]|type=2}; group1 packs data_size=2B,
// tensor_dim0 = tile_dim0 = nelem, tensor_dim1 = tile_dim1 = 1, no pad/iterate,
// workgroup_mask=0 (not in a cluster). Groups 2/3 unused (<=2D) -> zeros.
__device__ __forceinline__ void tdm_copy_f16(unsigned int lds_off, const void* gsrc,
                                             unsigned int nelem) {
  const unsigned long long ga = (unsigned long long)(uintptr_t)gsrc;
  u32x4 g0;
  g0[0] = 1u;                                                    // count=1 (user)
  g0[1] = lds_off;                                               // LDS byte address
  g0[2] = (unsigned int)ga;                                      // global_addr lo
  g0[3] = (unsigned int)((ga >> 32) & 0x01FFFFFFu) | (2u << 30); // addr hi | type=2
  i32x8 g1;
  g1[0] = (int)(1u << 16);                        // data_size = 2 bytes
  g1[1] = (int)((nelem & 0xFFFFu) << 16);         // tensor_dim0[15:0]
  g1[2] = (int)((nelem >> 16) | (1u << 16));      // tensor_dim0[31:16] | tensor_dim1=1
  g1[3] = (int)((nelem & 0xFFFFu) << 16);         // tensor_dim1[31:16]=0 | tile_dim0
  g1[4] = 1;                                      // tile_dim1=1, tile_dim2=0
  g1[5] = (int)nelem;                             // tensor_dim0_stride lo
  g1[6] = 0;                                      // stride0 hi | stride1 lo
  g1[7] = 0;                                      // stride1 hi
  i32x4 z4 = {0, 0, 0, 0};
#if defined(__clang_major__) && (__clang_major__ >= 23)
  i32x8 z8 = {0, 0, 0, 0, 0, 0, 0, 0};
  __builtin_amdgcn_tensor_load_to_lds(g0, g1, z4, z4, z8, 0);
#else
  __builtin_amdgcn_tensor_load_to_lds(g0, g1, z4, z4, 0);
#endif
}
#endif  // HAVE_TDM

// A-operand fragment: 16x32 tile, row-major LDS (rows stride `ld` halves).
// Lane 0-15: row=lane, K = {0..7, 16..23}; lane 16-31: row=lane-16, K = {8..15, 24..31}.
__device__ __forceinline__ h16 ld_afrag(const _Float16* base, int ld, int kofs, int lane) {
  int r  = lane & 15;
  int kb = (lane >> 4) << 3;  // 0 or 8
  const _Float16* p = base + r * ld + kofs + kb;
  h8 lo = *(const h8*)(p);
  h8 hi = *(const h8*)(p + 16);
  return __builtin_shufflevector(lo, hi, 0, 1, 2, 3, 4, 5, 6, 7,
                                 8, 9, 10, 11, 12, 13, 14, 15);
}

// B-operand fragment: 32x16 tile from n-major LDS (Bs[n*ld + k]).
// Lane 0-15: col=lane, K = {0..7, 16..23}; lane 16-31: col=lane-16, K = {8..15, 24..31}.
__device__ __forceinline__ h16 ld_bfrag(const _Float16* base, int ld, int kofs, int lane) {
  int n  = lane & 15;
  int kb = (lane >> 4) << 3;  // 0 or 8
  const _Float16* p = base + n * ld + kofs + kb;
  h8 lo = *(const h8*)(p);
  h8 hi = *(const h8*)(p + 16);
  return __builtin_shufflevector(lo, hi, 0, 1, 2, 3, 4, 5, 6, 7,
                                 8, 9, 10, 11, 12, 13, 14, 15);
}

// C[m,n] = sum_k A[m,k] * W[n,k] + bias[n], optionally scaled, M=8192 N=K=1024.
// Block: 128x128 tile, 256 threads (8 waves). Wave (wm,wn) owns 32x64 = 2x4 WMMA tiles.
// A_F16: A is f16 (ctx) vs f32 (inputs). HEADSPLIT: write f16 [B,H,S,DK] vs f32 [M,N].
template <bool A_F16, bool HEADSPLIT>
__global__ __launch_bounds__(256) void gemm_xwt_kernel(const void* __restrict__ Aptr,
                                                       const float* __restrict__ W,
                                                       const float* __restrict__ bias,
                                                       void* __restrict__ Out,
                                                       float out_scale) {
  __shared__ __align__(16) _Float16 As[128 * 32];  // row-major [m][k]
  __shared__ __align__(16) _Float16 Bs[128 * 32];  // n-major  [n][k]

  const int tid  = threadIdx.x;
  const int lane = tid & 31;
  const int wave = tid >> 5;
  const int wm   = wave >> 1;  // 0..3 -> 32-row strip
  const int wn   = wave & 1;   // 0..1 -> 64-col strip
  const int m0   = blockIdx.y * 128;
  const int n0   = blockIdx.x * 128;

  const int lrow = tid >> 1;        // 0..127
  const int lcol = (tid & 1) * 16;  // 0 or 16

  f8 acc[2][4];
#pragma unroll
  for (int mt = 0; mt < 2; ++mt)
#pragma unroll
    for (int nt = 0; nt < 4; ++nt) acc[mt][nt] = (f8)(0.0f);

  for (int k0 = 0; k0 < DMODEL; k0 += 32) {
    __syncthreads();
    // stage A tile (convert to f16 if needed)
    if (A_F16) {
      const _Float16* A = (const _Float16*)Aptr;
      const h8* src = (const h8*)(A + (size_t)(m0 + lrow) * DMODEL + k0 + lcol);
      *(h8*)(&As[lrow * 32 + lcol])     = src[0];
      *(h8*)(&As[lrow * 32 + lcol + 8]) = src[1];
    } else {
      const float* A = (const float*)Aptr;
      const float* src = A + (size_t)(m0 + lrow) * DMODEL + k0 + lcol;
#pragma unroll
      for (int j = 0; j < 16; ++j) As[lrow * 32 + lcol + j] = (_Float16)src[j];
    }
    // stage W tile, n-major (W is row-major [n][k] so this is a straight copy)
    {
      const float* src = W + (size_t)(n0 + lrow) * DMODEL + k0 + lcol;
#pragma unroll
      for (int j = 0; j < 16; ++j) Bs[lrow * 32 + lcol + j] = (_Float16)src[j];
    }
    // prefetch next k-slice while this one is consumed (-> global_prefetch_b8)
    if (k0 + 32 < DMODEL) {
      if (!A_F16)
        __builtin_prefetch((const float*)Aptr + (size_t)(m0 + lrow) * DMODEL + k0 + 32 + lcol);
      __builtin_prefetch(W + (size_t)(n0 + lrow) * DMODEL + k0 + 32 + lcol);
    }
    __syncthreads();

    h16 afr[2], bfr[4];
#pragma unroll
    for (int mt = 0; mt < 2; ++mt)
      afr[mt] = ld_afrag(&As[(wm * 32 + mt * 16) * 32], 32, 0, lane);
#pragma unroll
    for (int nt = 0; nt < 4; ++nt)
      bfr[nt] = ld_bfrag(&Bs[(wn * 64 + nt * 16) * 32], 32, 0, lane);
#pragma unroll
    for (int mt = 0; mt < 2; ++mt)
#pragma unroll
      for (int nt = 0; nt < 4; ++nt)
        acc[mt][nt] = wmma_16x16x32(afr[mt], bfr[nt], acc[mt][nt]);
  }

  // epilogue: + bias, * scale, store
  const int hi8 = (lane >> 4) * 8;
  const int cn  = lane & 15;
#pragma unroll
  for (int mt = 0; mt < 2; ++mt) {
#pragma unroll
    for (int nt = 0; nt < 4; ++nt) {
      const int n = n0 + wn * 64 + nt * 16 + cn;
      const float bv = bias[n];
#pragma unroll
      for (int r = 0; r < 8; ++r) {
        const int m = m0 + wm * 32 + mt * 16 + r + hi8;
        const float v = (acc[mt][nt][r] + bv) * out_scale;
        if (HEADSPLIT) {
          const int b = m >> 11, s = m & (SEQ - 1);
          const int h = n >> 6, d = n & (DK - 1);
          ((_Float16*)Out)[(((size_t)b * HEADS + h) * SEQ + s) * DK + d] = (_Float16)v;
        } else {
          ((float*)Out)[(size_t)m * DMODEL + n] = v;
        }
      }
    }
  }
}

// Flash attention: one block = 64 query rows of one (b,h); 4 waves x 16 rows.
// q is pre-scaled by 1/sqrt(dk). ctx written f16 as [B,S,DMODEL].
__global__ __launch_bounds__(128) void attn_kernel(const _Float16* __restrict__ q,
                                                   const _Float16* __restrict__ k,
                                                   const _Float16* __restrict__ v,
                                                   const int* __restrict__ mask,
                                                   _Float16* __restrict__ ctx) {
  __shared__ __align__(16) _Float16 Qs[64 * 64];      // [row][d]
  __shared__ __align__(16) _Float16 Ks[64 * 64];      // [token][d]  (n-major for QK^T)
  __shared__ __align__(16) _Float16 Vs[64 * 64];      // [d][token]  (n-major for P*V)
  __shared__ __align__(16) _Float16 Ps[4][16 * 64];   // per-wave P tile [row][token]

  const int tid  = threadIdx.x;
  const int lane = tid & 31;
  const int wid  = tid >> 5;  // 0..3
  const int b  = blockIdx.z;
  const int h  = blockIdx.y;
  const int q0 = blockIdx.x * 64;

  const _Float16* qbase = q + (((size_t)b * HEADS + h) * SEQ + q0) * DK;
  const _Float16* kbase = k + ((size_t)b * HEADS + h) * SEQ * DK;
  const _Float16* vbase = v + ((size_t)b * HEADS + h) * SEQ * DK;
  const int* mrow = mask + ((size_t)b * SEQ + q0) * SEQ;

  // stage resident Q tile (64x64 f16 = one contiguous 8KB chunk)
#if HAVE_TDM
  if (wid == 0) {
    tdm_copy_f16((unsigned int)(uintptr_t)&Qs[0], qbase, 64 * DK);
    __builtin_amdgcn_s_wait_tensorcnt(0);
  }
#else
  {
    const int row = tid >> 1, c0 = (tid & 1) * 32;
    const h8* src = (const h8*)(qbase + (size_t)row * DK + c0);
    h8* dst = (h8*)(&Qs[row * 64 + c0]);
#pragma unroll
    for (int j = 0; j < 4; ++j) dst[j] = src[j];
  }
#endif

  const int hi8 = (lane >> 4) * 8;
  const int cn  = lane & 15;

  float mstat[8], lstat[8];
  f8 oacc[4];
#pragma unroll
  for (int r = 0; r < 8; ++r) { mstat[r] = -3.0e38f; lstat[r] = 0.0f; }
#pragma unroll
  for (int ot = 0; ot < 4; ++ot) oacc[ot] = (f8)(0.0f);

  for (int kv0 = 0; kv0 < SEQ; kv0 += 64) {
    __syncthreads();  // previous iteration done reading Ks/Vs
    // K tile: contiguous 8KB -> TDM; V tile: transposed -> manual staging
#if HAVE_TDM
    if (wid == 0) tdm_copy_f16((unsigned int)(uintptr_t)&Ks[0],
                               kbase + (size_t)kv0 * DK, 64 * DK);
#else
    {
      const int row = tid >> 1, c0 = (tid & 1) * 32;
      const h8* ks = (const h8*)(kbase + (size_t)(kv0 + row) * DK + c0);
      h8* kd = (h8*)(&Ks[row * 64 + c0]);
#pragma unroll
      for (int j = 0; j < 4; ++j) kd[j] = ks[j];
    }
#endif
    {
      const int row = tid >> 1, c0 = (tid & 1) * 32;
      const _Float16* vs = vbase + (size_t)(kv0 + row) * DK + c0;
#pragma unroll
      for (int j = 0; j < 32; ++j) Vs[(c0 + j) * 64 + row] = vs[j];
    }
#if HAVE_TDM
    if (wid == 0) __builtin_amdgcn_s_wait_tensorcnt(0);
#endif
    __syncthreads();

    // scores: S = Qtile(16x64) x Ktile^T(64x64) -> 4 tiles of 16x16
    const h16 aq0 = ld_afrag(&Qs[(wid * 16) * 64], 64, 0, lane);
    const h16 aq1 = ld_afrag(&Qs[(wid * 16) * 64], 64, 32, lane);
    f8 st[4];
#pragma unroll
    for (int nt = 0; nt < 4; ++nt) {
      f8 c = (f8)(0.0f);
      c = wmma_16x16x32(aq0, ld_bfrag(&Ks[(nt * 16) * 64], 64, 0, lane), c);
      c = wmma_16x16x32(aq1, ld_bfrag(&Ks[(nt * 16) * 64], 64, 32, lane), c);
      st[nt] = c;
    }

    // mask + row max (reduce over 16 lanes of each half-wave)
    float rmax[8];
#pragma unroll
    for (int r = 0; r < 8; ++r) rmax[r] = -3.0e38f;
#pragma unroll
    for (int nt = 0; nt < 4; ++nt) {
      const int kg = kv0 + nt * 16 + cn;
#pragma unroll
      for (int r = 0; r < 8; ++r) {
        const int qr = wid * 16 + r + hi8;
        float s = st[nt][r];
        if (mrow[(size_t)qr * SEQ + kg] == 0) s = NEG_INF;
        st[nt][r] = s;
        rmax[r] = fmaxf(rmax[r], s);
      }
    }
#pragma unroll
    for (int r = 0; r < 8; ++r)
#pragma unroll
      for (int off = 1; off < 16; off <<= 1)
        rmax[r] = fmaxf(rmax[r], __shfl_xor(rmax[r], off, 32));

    // online softmax update
    float rsum[8];
#pragma unroll
    for (int r = 0; r < 8; ++r) {
      const float mnew = fmaxf(mstat[r], rmax[r]);
      const float corr = __expf(mstat[r] - mnew);
      mstat[r] = mnew;
      lstat[r] *= corr;
#pragma unroll
      for (int ot = 0; ot < 4; ++ot) oacc[ot][r] *= corr;
      rsum[r] = 0.0f;
    }
#pragma unroll
    for (int nt = 0; nt < 4; ++nt)
#pragma unroll
      for (int r = 0; r < 8; ++r) {
        const float p = __expf(st[nt][r] - mstat[r]);
        st[nt][r] = p;
        rsum[r] += p;
      }
#pragma unroll
    for (int r = 0; r < 8; ++r) {
#pragma unroll
      for (int off = 1; off < 16; off <<= 1) rsum[r] += __shfl_xor(rsum[r], off, 32);
      lstat[r] += rsum[r];
    }

    // C-layout -> A-layout for P via per-wave LDS buffer
#pragma unroll
    for (int nt = 0; nt < 4; ++nt)
#pragma unroll
      for (int r = 0; r < 8; ++r)
        Ps[wid][(r + hi8) * 64 + nt * 16 + cn] = (_Float16)st[nt][r];
    __syncthreads();

    // O += P(16x64) x Vtile(64x64)
    const h16 ap0 = ld_afrag(&Ps[wid][0], 64, 0, lane);
    const h16 ap1 = ld_afrag(&Ps[wid][0], 64, 32, lane);
#pragma unroll
    for (int ot = 0; ot < 4; ++ot) {
      oacc[ot] = wmma_16x16x32(ap0, ld_bfrag(&Vs[(ot * 16) * 64], 64, 0, lane), oacc[ot]);
      oacc[ot] = wmma_16x16x32(ap1, ld_bfrag(&Vs[(ot * 16) * 64], 64, 32, lane), oacc[ot]);
    }
  }

  // normalize and store ctx[b][s][h*64+d] (f16, merged-head layout)
#pragma unroll
  for (int ot = 0; ot < 4; ++ot)
#pragma unroll
    for (int r = 0; r < 8; ++r) {
      const float o = oacc[ot][r] / lstat[r];
      const int qr = q0 + wid * 16 + r + hi8;
      const int d  = h * DK + ot * 16 + cn;
      ctx[((size_t)b * SEQ + qr) * DMODEL + d] = (_Float16)o;
    }
}

extern "C" void kernel_launch(void* const* d_in, const int* in_sizes, int n_in,
                              void* d_out, int out_size, void* d_ws, size_t ws_size,
                              hipStream_t stream) {
  (void)in_sizes; (void)n_in; (void)out_size; (void)ws_size;
  const float* Q    = (const float*)d_in[0];
  const float* K    = (const float*)d_in[1];
  const float* V    = (const float*)d_in[2];
  const int*   mask = (const int*)d_in[3];
  const float* Wq = (const float*)d_in[4];
  const float* bq = (const float*)d_in[5];
  const float* Wk = (const float*)d_in[6];
  const float* bk = (const float*)d_in[7];
  const float* Wv = (const float*)d_in[8];
  const float* bv = (const float*)d_in[9];
  const float* Wo = (const float*)d_in[10];
  const float* bo = (const float*)d_in[11];

  // workspace carve: 4 x (8192*1024) f16 buffers = 64 MiB
  const size_t NELEM = (size_t)MTOT * DMODEL;  // 8388608
  _Float16* q16  = (_Float16*)d_ws;
  _Float16* k16  = q16 + NELEM;
  _Float16* v16  = k16 + NELEM;
  _Float16* ctx  = v16 + NELEM;

  dim3 ggemm(DMODEL / 128, MTOT / 128, 1);  // (8, 64)
  const float qscale = 0.125f;              // 1/sqrt(64)

  gemm_xwt_kernel<false, true><<<ggemm, 256, 0, stream>>>(Q, Wq, bq, q16, qscale);
  gemm_xwt_kernel<false, true><<<ggemm, 256, 0, stream>>>(K, Wk, bk, k16, 1.0f);
  gemm_xwt_kernel<false, true><<<ggemm, 256, 0, stream>>>(V, Wv, bv, v16, 1.0f);

  dim3 gattn(SEQ / 64, HEADS, BATCH);  // (32, 16, 4)
  attn_kernel<<<gattn, 128, 0, stream>>>(q16, k16, v16, mask, ctx);

  gemm_xwt_kernel<true, false><<<ggemm, 256, 0, stream>>>(ctx, Wo, bo, d_out, 1.0f);
}
